// MyBiGRU_crf_49486613184878
// MI455X (gfx1250) — matrix operand order repaired and checked by reading, not compiled, not run on hardware
//
#include <hip/hip_runtime.h>
#include <math.h>

// ---------------- problem constants (fixed by reference) ----------------
constexpr int V = 8000, E = 100, H = 64, K = 17;
constexpr int B = 256, T = 512;
constexpr int G4H = 4 * H;   // 256 gate columns
constexpr int KP  = 128;     // E padded to 128 for K=32 WMMA steps

typedef _Float16 half_t;
typedef _Float16 v16h __attribute__((ext_vector_type(16)));
typedef float    v8f  __attribute__((ext_vector_type(8)));

// fast sigmoid: one v_exp + one v_rcp (no IEEE divide expansion)
__device__ __forceinline__ float fast_sigmoid(float x) {
  return __builtin_amdgcn_rcpf(1.f + __expf(-x));
}

// Lightweight workgroup barrier for the recurrence: only LDS (DScnt) ordering is
// required -- do NOT drain loadcnt/storecnt, so prefetched global loads and hseq
// stores stay in flight across the barrier (unlike __syncthreads()'s full fence).
__device__ __forceinline__ void lds_barrier() {
  asm volatile("s_wait_dscnt 0x0\n\t"
               "s_barrier_signal -1\n\t"
               "s_barrier_wait -1" ::: "memory");
}

// ---------------- kernel 1: weight conversion / transpose / pad / gate-interleave ----------------
// Column permutation: physical column p = j*4 + gate  (j = hidden index, gate: 0=i 1=f 2=g 3=o).
// Original PyTorch row n_orig = gate*64 + j. A 16-wide WMMA N-tile then holds all 4 gates
// for 4 hidden indices -> c/h update needs only lane gathers within groups of 4.
__global__ void crf_prep_kernel(const float* __restrict__ Wih_f, const float* __restrict__ Whh_f,
                                const float* __restrict__ bih_f, const float* __restrict__ bhh_f,
                                const float* __restrict__ Wih_b, const float* __restrict__ Whh_b,
                                const float* __restrict__ bih_b, const float* __restrict__ bhh_b,
                                half_t* __restrict__ WihT, half_t* __restrict__ WhhT,
                                float* __restrict__ bsum) {
  int idx = blockIdx.x * blockDim.x + threadIdx.x;
  if (idx < 2 * KP * G4H) {            // WihT[d][k][p], k zero-padded 100..127
    int d = idx / (KP * G4H), r = idx % (KP * G4H);
    int k = r / G4H, p = r % G4H;
    int n_orig = (p & 3) * 64 + (p >> 2);
    const float* W = d ? Wih_b : Wih_f;            // [4H, E]
    float v = (k < E) ? W[n_orig * E + k] : 0.f;
    WihT[idx] = (half_t)v;
  }
  if (idx < 2 * H * G4H) {             // WhhT[d][k][p]
    int d = idx / (H * G4H), r = idx % (H * G4H);
    int k = r / G4H, p = r % G4H;
    int n_orig = (p & 3) * 64 + (p >> 2);
    const float* W = d ? Whh_b : Whh_f;            // [4H, H]
    WhhT[idx] = (half_t)W[n_orig * H + k];
  }
  if (idx < 2 * G4H) {                 // bsum[d][p] = bih + bhh (permuted)
    int d = idx / G4H, p = idx % G4H;
    int n_orig = (p & 3) * 64 + (p >> 2);
    bsum[idx] = d ? (bih_b[n_orig] + bhh_b[n_orig]) : (bih_f[n_orig] + bhh_f[n_orig]);
  }
}

// ---------------- kernel 2: embedding gather -> padded f16 rows ----------------
__global__ void crf_gather_kernel(const int* __restrict__ data, const float* __restrict__ emb,
                                  half_t* __restrict__ xpad) {
  int bt = blockIdx.x;            // b*T + t
  int k  = threadIdx.x;           // 0..127
  int tok = data[bt];
  float v = (k < E) ? emb[(size_t)tok * E + k] : 0.f;
  xpad[(size_t)bt * KP + k] = (half_t)v;
}

// ---------------- kernel 3: fused BiLSTM recurrence (WMMA) ----------------
// grid = 32 blocks: blockIdx.x = dir*16 + batch_tile. block = 512 threads = 16 waves.
// Each wave owns 16 gate columns (4 hidden indices x 4 gates); M=16 batch rows per block.
// Weight B-fragments live in registers. x A-fragments are software-pipelined one step
// ahead; the light DS-only barrier keeps those loads in flight across timesteps.
__global__ __launch_bounds__(512) void crf_lstm_kernel(const half_t* __restrict__ xpad,
                                                       const half_t* __restrict__ WihT,
                                                       const half_t* __restrict__ WhhT,
                                                       const float* __restrict__ bsum,
                                                       half_t* __restrict__ hseq) {
  const int dir   = blockIdx.x >> 4;
  const int bbase = (blockIdx.x & 15) * 16;
  const int tid   = threadIdx.x;
  const int wave  = tid >> 5, lane = tid & 31;
  const int mlo   = lane & 15, hi = lane >> 4;   // A-frag row / K-half select
  const int nbase = wave * 16;
  const int p     = nbase + mlo;                 // permuted gate column for this lane
  const int g     = p & 3;                       // gate id: 0=i 1=f 2=g 3=o
  const int j     = p >> 2;                      // hidden index
  const int base  = lane & ~3;                   // lane group base for gathers

  __shared__ half_t sh[2][16 * H];               // double-buffered h state, 4 KB total

  for (int i = tid; i < 16 * H; i += 512) sh[0][i] = (half_t)0.f;
  __syncthreads();

  // ---- hoist loop-invariant weight B-fragments into registers ----
  const half_t* WihTd = WihT + (size_t)dir * KP * G4H;
  const half_t* WhhTd = WhhT + (size_t)dir * H * G4H;
  v16h bwih[4], bwhh[2];
#pragma unroll
  for (int kk = 0; kk < 4; ++kk)
#pragma unroll
    for (int e = 0; e < 16; ++e)
      bwih[kk][e] = WihTd[(size_t)(kk * 32 + (e >> 3) * 16 + hi * 8 + (e & 7)) * G4H + p];
#pragma unroll
  for (int kk = 0; kk < 2; ++kk)
#pragma unroll
    for (int e = 0; e < 16; ++e)
      bwhh[kk][e] = WhhTd[(size_t)(kk * 32 + (e >> 3) * 16 + hi * 8 + (e & 7)) * G4H + p];

  const float biasv = bsum[dir * G4H + p];
  const bool  is_g  = (g == 2);                  // loop-invariant lane predicate
  float creg[8];                                 // cell state, replicated in lane group of 4
#pragma unroll
  for (int r = 0; r < 8; ++r) creg[r] = 0.f;

  // x A-fragment loader (two contiguous 16B groups per K-step -> b128 loads)
  const half_t* xbase = xpad + (size_t)(bbase + mlo) * T * KP;
  auto load_x = [&](int t, v16h* dst) {
    const half_t* xrow = xbase + (size_t)t * KP;
#pragma unroll
    for (int kk = 0; kk < 4; ++kk) {
      const int k0 = kk * 32;
#pragma unroll
      for (int e = 0; e < 8; ++e) {
        dst[kk][e]     = xrow[k0 + hi * 8 + e];
        dst[kk][8 + e] = xrow[k0 + 16 + hi * 8 + e];
      }
    }
  };

  // hseq store base for this lane's (hi, j); row r adds a compile-time offset r*T*H
  half_t* hq0 = hseq + (((size_t)dir * B + bbase + 8 * hi) * T) * H + j;

  // one fused LSTM timestep: consumes axc (preloaded), prefetches step s+1 into axn
  auto lstm_step = [&](int s, v16h* axc, v16h* axn) {
    const int t = dir ? (T - 1 - s) : s;
    const half_t* shr = sh[s & 1];
    half_t*       shw = sh[(s + 1) & 1];

    // recurrent A-fragments from LDS (fresh after the barrier)
    v16h ah[2];
#pragma unroll
    for (int kk = 0; kk < 2; ++kk) {
      const int k0 = kk * 32;
#pragma unroll
      for (int e = 0; e < 8; ++e) {
        ah[kk][e]     = shr[mlo * H + k0 + hi * 8 + e];
        ah[kk][8 + e] = shr[mlo * H + k0 + 16 + hi * 8 + e];
      }
    }

    // two independent accumulator chains (depth 4 / depth 2) pipeline in the XDL unit
    v8f accx, acch;
#pragma unroll
    for (int r = 0; r < 8; ++r) { accx[r] = biasv; acch[r] = 0.f; }
#pragma unroll
    for (int kk = 0; kk < 2; ++kk)
      acch = __builtin_amdgcn_wmma_f32_16x16x32_f16(false, ah[kk], false, bwhh[kk], (short)0, acch, false, false);
#pragma unroll
    for (int kk = 0; kk < 4; ++kk)
      accx = __builtin_amdgcn_wmma_f32_16x16x32_f16(false, axc[kk], false, bwih[kk], (short)0, accx, false, false);

    // prefetch x for the NEXT timestep; latency hides behind activations + barrier
    const int sn = (s + 1 < T) ? (s + 1) : s;
    load_x(dir ? (T - 1 - sn) : sn, axn);

    half_t* shwq = shw + (8 * hi) * H + j;
    half_t* hq   = hq0 + (size_t)t * H;

    // activations in groups of 4 with phase-split loops -> 4 independent
    // v_exp/v_rcp chains interleave instead of serializing on TRANS latency
#pragma unroll
    for (int rg = 0; rg < 2; ++rg) {
      float act4[4];
#pragma unroll
      for (int q = 0; q < 4; ++q) {              // phase 1: gate activations
        const int r = rg * 4 + q;
        float gr  = accx[r] + acch[r];
        float xs  = is_g ? 2.f * gr : gr;        // tanh(x) = 2*sigmoid(2x) - 1
        float sg  = fast_sigmoid(xs);
        act4[q] = is_g ? 2.f * sg - 1.f : sg;
      }
      float iv4[4], fv4[4], gv4[4], ov4[4];
#pragma unroll
      for (int q = 0; q < 4; ++q) {              // phase 2: group-of-4 gathers
        iv4[q] = __shfl(act4[q], base + 0, 32);
        fv4[q] = __shfl(act4[q], base + 1, 32);
        gv4[q] = __shfl(act4[q], base + 2, 32);
        ov4[q] = __shfl(act4[q], base + 3, 32);
      }
#pragma unroll
      for (int q = 0; q < 4; ++q) {              // phase 3: c/h update + stores
        const int r = rg * 4 + q;
        float c = fv4[q] * creg[r] + iv4[q] * gv4[q];
        creg[r] = c;
        float th = 2.f * fast_sigmoid(2.f * c) - 1.f;   // tanh(c)
        float h  = ov4[q] * th;
        if (g == 0) {                            // one writer per group of 4 (stores only)
          shwq[r * H]            = (half_t)h;    // immediate-offset DS stores
          hq[(size_t)r * T * H]  = (half_t)h;    // immediate-offset global stores
        }
      }
    }
    lds_barrier();                               // DS-only wait: prefetch stays in flight
  };

  v16h axA[4], axB[4];
  load_x(dir ? (T - 1) : 0, axA);                // prologue: preload step 0

  for (int s = 0; s < T; s += 2) {               // 2x unroll to ping-pong ax buffers
    lstm_step(s,     axA, axB);
    lstm_step(s + 1, axB, axA);
  }
}

// ---------------- kernel 4: emissions h[2H] @ W_out^T + b_out ----------------
__global__ void crf_emission_kernel(const half_t* __restrict__ hseq, const float* __restrict__ W_out,
                                    const float* __restrict__ b_out, float* __restrict__ emis) {
  int bt = blockIdx.x;
  int k  = threadIdx.x;
  if (k >= K) return;
  const half_t* hf = hseq + (size_t)bt * H;
  const half_t* hb = hseq + ((size_t)B * T + bt) * H;
  float acc = b_out[k];
  for (int jj = 0; jj < H; ++jj) {
    acc += (float)hf[jj] * W_out[k * 2 * H + jj];
    acc += (float)hb[jj] * W_out[k * 2 * H + H + jj];
  }
  emis[(size_t)bt * K + k] = acc;
}

// ---------------- kernel 5: Viterbi forward + backtrack ----------------
__global__ void crf_viterbi_kernel(const float* __restrict__ emis, const unsigned char* __restrict__ mask,
                                   const float* __restrict__ trans, const float* __restrict__ start_trans,
                                   const float* __restrict__ end_trans, int* __restrict__ bp,
                                   float* __restrict__ out) {
  int b = blockIdx.x;
  int j = threadIdx.x;          // lane j owns tag j (j < K)
  __shared__ float ss[K];
  __shared__ float strans[K * K];
  for (int i = j; i < K * K; i += 32) strans[i] = trans[i];
  float score = 0.f;
  if (j < K) score = start_trans[j] + emis[((size_t)b * T) * K + j];
  __syncthreads();

  for (int t = 1; t < T; ++t) {
    if (j < K) ss[j] = score;
    __syncthreads();
    if (j < K) {
      float em = emis[((size_t)b * T + t) * K + j];
      float best = -3.4e38f; int bi = 0;
      for (int i = 0; i < K; ++i) {
        float c = ss[i] + strans[i * K + j] + em;
        if (c > best) { best = c; bi = i; }
      }
      bool mt = mask[b * T + t] != 0;            // all-ones in reference
      bp[((size_t)b * (T - 1) + (t - 1)) * K + j] = mt ? bi : j;
      score = mt ? best : score;
    }
    __syncthreads();
  }

  if (j < K) ss[j] = score + end_trans[j];
  __syncthreads();
  if (j == 0) {
    float best = -3.4e38f; int last = 0;
    for (int i = 0; i < K; ++i) if (ss[i] > best) { best = ss[i]; last = i; }
    out[b] = best;                                // best_score[B]
    float* paths = out + B;                       // paths[B,T] as float
    int tag = last;
    paths[(size_t)b * T + (T - 1)] = (float)tag;
    for (int t = T - 2; t >= 0; --t) {
      tag = bp[((size_t)b * (T - 1) + t) * K + tag];
      paths[(size_t)b * T + t] = (float)tag;
    }
  }
}

// ---------------- launcher ----------------
extern "C" void kernel_launch(void* const* d_in, const int* in_sizes, int n_in,
                              void* d_out, int out_size, void* d_ws, size_t ws_size,
                              hipStream_t stream) {
  (void)in_sizes; (void)n_in; (void)out_size; (void)ws_size;
  const int*           data  = (const int*)d_in[0];
  const unsigned char* mask  = (const unsigned char*)d_in[1];   // jnp bool -> bytes
  const float* emb   = (const float*)d_in[2];
  const float* Wih_f = (const float*)d_in[3];
  const float* Whh_f = (const float*)d_in[4];
  const float* bih_f = (const float*)d_in[5];
  const float* bhh_f = (const float*)d_in[6];
  const float* Wih_b = (const float*)d_in[7];
  const float* Whh_b = (const float*)d_in[8];
  const float* bih_b = (const float*)d_in[9];
  const float* bhh_b = (const float*)d_in[10];
  const float* W_out = (const float*)d_in[11];
  const float* b_out = (const float*)d_in[12];
  const float* trans = (const float*)d_in[13];
  const float* start_trans = (const float*)d_in[14];
  const float* end_trans   = (const float*)d_in[15];

  char*  ws  = (char*)d_ws;
  size_t off = 0;
  auto alloc = [&](size_t bytes) -> void* {
    void* pp = ws + off;
    off = (off + bytes + 255) & ~(size_t)255;
    return pp;
  };
  half_t* xpad = (half_t*)alloc((size_t)B * T * KP * sizeof(half_t));     // 33.5 MB (fits L2)
  half_t* hseq = (half_t*)alloc((size_t)2 * B * T * H * sizeof(half_t));  // 33.5 MB
  half_t* WihT = (half_t*)alloc((size_t)2 * KP * G4H * sizeof(half_t));
  half_t* WhhT = (half_t*)alloc((size_t)2 * H * G4H * sizeof(half_t));
  float*  bsum = (float*)alloc((size_t)2 * G4H * sizeof(float));
  float*  emis = (float*)alloc((size_t)B * T * K * sizeof(float));        // 8.9 MB
  int*    bp   = (int*)alloc((size_t)B * (T - 1) * K * sizeof(int));      // 8.9 MB

  crf_prep_kernel<<<(2 * KP * G4H + 255) / 256, 256, 0, stream>>>(
      Wih_f, Whh_f, bih_f, bhh_f, Wih_b, Whh_b, bih_b, bhh_b, WihT, WhhT, bsum);
  crf_gather_kernel<<<B * T, KP, 0, stream>>>(data, emb, xpad);
  crf_lstm_kernel<<<32, 512, 0, stream>>>(xpad, WihT, WhhT, bsum, hseq);
  crf_emission_kernel<<<B * T, 32, 0, stream>>>(hseq, W_out, b_out, emis);
  crf_viterbi_kernel<<<B, 32, 0, stream>>>(emis, mask, trans, start_trans, end_trans, bp,
                                           (float*)d_out);
}
